// CrossLayer_15745350107627
// MI455X (gfx1250) — matrix operand order repaired and checked by reading, not compiled, not run on hardware
//
#include <hip/hip_runtime.h>
#include <hip/hip_bf16.h>

typedef __attribute__((ext_vector_type(16))) __bf16 v16bf;
typedef __attribute__((ext_vector_type(16))) short  v16s;
typedef __attribute__((ext_vector_type(8)))  float  v8f;
typedef __attribute__((ext_vector_type(4)))  int    i4v;
typedef __attribute__((ext_vector_type(4)))  unsigned int u32x4;
typedef __attribute__((ext_vector_type(8)))  int    i32x8;
typedef __attribute__((ext_vector_type(4)))  int    i32x4;

#define B_   2
#define N_   2048
#define DIM  1024
#define HID  4096
#define NH   16
#define HD   64
#define TOK  (B_ * N_)

__device__ __forceinline__ unsigned short f2bf(float f) {
  unsigned u = __builtin_bit_cast(unsigned, f);
  u += 0x7FFFu + ((u >> 16) & 1u);
  return (unsigned short)(u >> 16);
}

// ---------------------------------------------------------------------------
// fp32 -> bf16 (grid-stride)
// ---------------------------------------------------------------------------
__global__ void cvt_bf16_k(const float* __restrict__ in,
                           unsigned short* __restrict__ out, int n) {
  for (int i = blockIdx.x * blockDim.x + threadIdx.x; i < n;
       i += gridDim.x * blockDim.x)
    out[i] = f2bf(in[i]);
}

// fp32 [K,N] -> bf16 transposed [N,K]
__global__ void cvt_tr_k(const float* __restrict__ in,
                         unsigned short* __restrict__ out, int K, int N) {
  for (int i = blockIdx.x * blockDim.x + threadIdx.x; i < K * N;
       i += gridDim.x * blockDim.x) {
    const int k = i / N, n = i - k * N;
    out[(size_t)n * K + k] = f2bf(in[i]);
  }
}

// ---------------------------------------------------------------------------
// RMSNorm over rows of DIM, bf16 out
// ---------------------------------------------------------------------------
__global__ __launch_bounds__(256)
void rmsnorm_rows_k(const float* __restrict__ X, const float* __restrict__ W,
                    unsigned short* __restrict__ Y) {
  __shared__ float red[256];
  const int row = blockIdx.x;
  const float* xr = X + (size_t)row * DIM;
  const int c0 = threadIdx.x * 4;
  float xl[4], s = 0.f;
#pragma unroll
  for (int j = 0; j < 4; ++j) { xl[j] = xr[c0 + j]; s += xl[j] * xl[j]; }
  red[threadIdx.x] = s;
  __syncthreads();
  for (int st = 128; st > 0; st >>= 1) {
    if (threadIdx.x < st) red[threadIdx.x] += red[threadIdx.x + st];
    __syncthreads();
  }
  const float scale = rsqrtf(red[0] / (float)DIM + 1e-6f);
#pragma unroll
  for (int j = 0; j < 4; ++j)
    Y[(size_t)row * DIM + c0 + j] = f2bf(xl[j] * scale * W[c0 + j]);
}

// ---------------------------------------------------------------------------
// NT GEMM: C[M,N] = A[M,K] * Bt[N,K]^T  (bf16 in, fp32 acc, WMMA)
// A tile stored row-major in LDS (A-frag contiguous); B tile DMA'd by the TDM.
// mode 0: Cf=acc  1: Cb=bf16(acc)  2: Cf+=acc  3: Cb=bf16(silu(Cf)*acc)
// ---------------------------------------------------------------------------
__global__ __launch_bounds__(256)
void gemm_bf16_k(const unsigned short* __restrict__ A,
                 const unsigned short* __restrict__ Bt,
                 float* __restrict__ Cf, unsigned short* __restrict__ Cb,
                 int M, int N, int K, int mode) {
  __shared__ __attribute__((aligned(32))) unsigned short Asw[128 * 32];
  __shared__ __attribute__((aligned(32))) unsigned short Btile[64 * 32];
  const int tid = threadIdx.x;
  const int lane = tid & 31;
  const int w  = tid >> 5;
  const int wm = w & 3;
  const int wn = w >> 2;
  const int bm = blockIdx.y * 128;
  const int bn = blockIdx.x * 64;
  const int hv = lane >> 4;
  const int ln = lane & 15;

  v8f c[2][2] = {};

  const int arow = tid >> 1;            // 0..127
  const int acol = (tid & 1) * 16;      // 0 or 16

  const unsigned ldsB = (unsigned)(uintptr_t)(void*)&Btile[0];

  for (int k0 = 0; k0 < K; k0 += 32) {
    // ---- TDM: B tile 64 rows x 32 k, row-major from Bt[N,K] ----
    if (w == 0) {
      const unsigned long long ga =
          (unsigned long long)(uintptr_t)(Bt + (size_t)bn * K + k0);
      u32x4 g0;
      g0[0] = 1u;                                   // count=1 valid descriptor
      g0[1] = ldsB;                                 // lds_addr
      g0[2] = (unsigned)(ga & 0xFFFFFFFFu);         // global_addr lo
      g0[3] = (unsigned)((ga >> 32) & 0x01FFFFFFu) | (2u << 30);  // hi | type=2
      i32x8 g1;
      const unsigned td0 = (unsigned)K, td1 = (unsigned)N;
      g1[0] = (int)(1u << 16);                          // data_size = 2 bytes
      g1[1] = (int)((td0 & 0xFFFFu) << 16);             // tensor_dim0 lo
      g1[2] = (int)((td0 >> 16) | ((td1 & 0xFFFFu) << 16));
      g1[3] = (int)((td1 >> 16) | (32u << 16));         // tile_dim0 = 32
      g1[4] = (int)64u;                                 // tile_dim1 = 64
      g1[5] = (int)(unsigned)K;                         // dim0 stride lo
      g1[6] = 0;
      g1[7] = 0;
      const i32x4 gz4 = {0, 0, 0, 0};
      const i32x8 gz8 = {0, 0, 0, 0, 0, 0, 0, 0};
      __builtin_amdgcn_tensor_load_to_lds(g0, g1, gz4, gz4, gz8, 0);
    }

    // ---- A tile, row-major (contiguous k per 32-K block) ----
    {
      const unsigned short* srcp = A + (size_t)(bm + arow) * K + k0 + acol;
      if (k0 + 32 < K) __builtin_prefetch(srcp + 32, 0, 1);
      const i4v lo = *(const i4v*)(srcp);       // 8 bf16
      const i4v hi = *(const i4v*)(srcp + 8);   // 8 bf16
      *(i4v*)(Asw + arow * 32 + acol)  = lo;
      *(i4v*)(Asw + arow * 32 + acol + 8) = hi;
    }

    if (w == 0) __builtin_amdgcn_s_wait_tensorcnt(0);
    __syncthreads();

    v16bf af[2], bfr[2];
#pragma unroll
    for (int fr = 0; fr < 2; ++fr) {
      const int row = wm * 32 + fr * 16 + ln;
      af[fr] = __builtin_bit_cast(
          v16bf, *(const v16s*)(Asw + row * 32 + hv * 16));
    }
#pragma unroll
    for (int fc = 0; fc < 2; ++fc) {
      const int col = wn * 32 + fc * 16 + ln;
      bfr[fc] = __builtin_bit_cast(
          v16bf, *(const v16s*)(Btile + col * 32 + hv * 16));
    }
#pragma unroll
    for (int fr = 0; fr < 2; ++fr)
#pragma unroll
      for (int fc = 0; fc < 2; ++fc)
        c[fr][fc] = __builtin_amdgcn_wmma_f32_16x16x32_bf16(
            false, af[fr], false, bfr[fc], (short)0, c[fr][fc], false, false);
    __syncthreads();
  }

#pragma unroll
  for (int fr = 0; fr < 2; ++fr)
#pragma unroll
    for (int fc = 0; fc < 2; ++fc)
#pragma unroll
      for (int r = 0; r < 8; ++r) {
        const int row = bm + wm * 32 + fr * 16 + hv * 8 + r;
        const int col = bn + wn * 32 + fc * 16 + ln;
        const size_t idx = (size_t)row * N + col;
        const float v = c[fr][fc][r];
        if (mode == 0)      Cf[idx] = v;
        else if (mode == 1) Cb[idx] = f2bf(v);
        else if (mode == 2) Cf[idx] += v;
        else {
          const float z = Cf[idx];
          Cb[idx] = f2bf((z / (1.f + __expf(-z))) * v);
        }
      }
}

// ---------------------------------------------------------------------------
// per-head RMSNorm + RoPE; raw [TOK, rstride] fp32 -> [b*h, N_, 64] bf16
// swz=1 applies the A-fragment K-group swizzle (unused: Q and K pass swz=0)
// ---------------------------------------------------------------------------
__global__ __launch_bounds__(256)
void prep_qk_k(const float* __restrict__ R, int rstride,
               const float* __restrict__ nw, const int* __restrict__ pos,
               unsigned short* __restrict__ O, int swz) {
  __shared__ float xs[DIM];
  __shared__ float ys[DIM];
  __shared__ float part[256];
  __shared__ float sc[NH];
  const int tok = blockIdx.x;
  const int bb  = tok / N_;
  const int row = tok - bb * N_;
  const int c0  = threadIdx.x * 4;
  float ssum = 0.f;
#pragma unroll
  for (int j = 0; j < 4; ++j) {
    const float v = R[(size_t)tok * rstride + c0 + j];
    xs[c0 + j] = v;
    ssum += v * v;
  }
  part[threadIdx.x] = ssum;
  __syncthreads();
  if (threadIdx.x < NH) {
    float s = 0.f;
    for (int j = 0; j < 16; ++j) s += part[threadIdx.x * 16 + j];
    sc[threadIdx.x] = rsqrtf(s / (float)HD + 1e-6f);
  }
  __syncthreads();
  const int head = c0 >> 6;
#pragma unroll
  for (int j = 0; j < 4; ++j) {
    const int cc = c0 + j;
    ys[cc] = xs[cc] * sc[head] * nw[cc & 63];
  }
  __syncthreads();
  const float p = (float)pos[tok];
#pragma unroll
  for (int j = 0; j < 4; ++j) {
    const int cc = c0 + j;
    const int dd = cc & 63;
    const int jr = dd & 31;
    const float inv = __expf(-(2.f * (float)jr / (float)HD) * 9.210340371976184f);
    float sn, cs;
    __sincosf(p * inv, &sn, &cs);
    const float v = (dd < 32) ? (ys[cc] * cs - ys[cc + 32] * sn)
                              : (ys[cc] * cs + ys[cc - 32] * sn);
    int dp = dd;
    if (swz) {
      const int r5 = dd & 31, g = r5 >> 3;
      const int pin = (g == 1) ? r5 + 8 : (g == 2) ? r5 - 8 : r5;
      dp = (dd & 32) + pin;
    }
    O[((size_t)(bb * NH + head) * N_ + row) * HD + dp] = f2bf(v);
  }
}

// ---------------------------------------------------------------------------
// V: kv_raw [TOK, 2*DIM] fp32 -> Vt [b*h, 64(feat), N_(key)] bf16 (transposed)
// ---------------------------------------------------------------------------
__global__ void cvt_v_k(const float* __restrict__ KV,
                        unsigned short* __restrict__ Vt) {
  const int i = blockIdx.x * blockDim.x + threadIdx.x;
  if (i >= TOK * DIM) return;
  const int tok = i / DIM;
  const int cc = i - tok * DIM;
  const int bb = tok / N_, row = tok - bb * N_;
  const int head = cc >> 6, dd = cc & 63;
  Vt[((size_t)(bb * NH + head) * HD + dd) * N_ + row] =
      f2bf(KV[(size_t)tok * (2 * DIM) + DIM + cc]);
}

// ---------------------------------------------------------------------------
// Flash attention. Q [bh,N_,64]; K [bh,N_,64]; Vt [bh,64,N_].
// grid (N_/128, B_*NH); each wave owns 16 q rows, 32 keys/step.
// ---------------------------------------------------------------------------
__global__ __launch_bounds__(256)
void flash_attn_k(const unsigned short* __restrict__ Q,
                  const unsigned short* __restrict__ Kb,
                  const unsigned short* __restrict__ Vt,
                  unsigned short* __restrict__ O) {
  __shared__ __attribute__((aligned(32))) unsigned short Pst[8 * 16 * 32];
  const int tid = threadIdx.x, lane = tid & 31, w = tid >> 5;
  const int bh = blockIdx.y;
  const int bb = bh / NH, head = bh - bb * NH;
  const int q0 = blockIdx.x * 128 + w * 16;
  const int hv = lane >> 4, ln = lane & 15;
  const size_t base = (size_t)bh * N_ * HD;
  unsigned short* Pw = Pst + w * 16 * 32;

  v16bf qa[2];
  {
    const int row = q0 + ln;
#pragma unroll
    for (int ks = 0; ks < 2; ++ks)
      qa[ks] = __builtin_bit_cast(
          v16bf, *(const v16s*)(Q + base + (size_t)row * HD + ks * 32 + hv * 16));
  }

  float mrow[8], lrow[8];
  v8f o[4] = {};
#pragma unroll
  for (int r = 0; r < 8; ++r) { mrow[r] = -1e30f; lrow[r] = 0.f; }

  for (int m0 = 0; m0 < N_; m0 += 32) {
    v8f s[2] = {};
#pragma unroll
    for (int nt = 0; nt < 2; ++nt)
#pragma unroll
      for (int ks = 0; ks < 2; ++ks) {
        const int key = m0 + nt * 16 + ln;
        const v16bf kf = __builtin_bit_cast(
            v16bf,
            *(const v16s*)(Kb + base + (size_t)key * HD + ks * 32 + hv * 16));
        s[nt] = __builtin_amdgcn_wmma_f32_16x16x32_bf16(
            false, qa[ks], false, kf, (short)0, s[nt], false, false);
      }

    float corr[8];
#pragma unroll
    for (int r = 0; r < 8; ++r) {
      const float a = s[0][r] * 0.125f;
      const float b2 = s[1][r] * 0.125f;
      float v = fmaxf(a, b2);
#pragma unroll
      for (int off = 1; off < 16; off <<= 1)
        v = fmaxf(v, __shfl_xor(v, off, 32));
      const float mnew = fmaxf(mrow[r], v);
      corr[r] = __expf(mrow[r] - mnew);
      const float p0 = __expf(a - mnew), p1 = __expf(b2 - mnew);
      s[0][r] = p0;
      s[1][r] = p1;
      float ps = p0 + p1;
#pragma unroll
      for (int off = 1; off < 16; off <<= 1) ps += __shfl_xor(ps, off, 32);
      lrow[r] = lrow[r] * corr[r] + ps;
      mrow[r] = mnew;
    }

    // stage P row-major (contiguous k) so the A-frag read below is one contiguous 32B load
#pragma unroll
    for (int r = 0; r < 8; ++r) {
      const int row = hv * 8 + r;
      Pw[row * 32 + ln]       = f2bf(s[0][r]);
      Pw[row * 32 + 16 + ln]  = f2bf(s[1][r]);
    }

    const v16bf pa = __builtin_bit_cast(
        v16bf, *(const v16s*)(Pw + ln * 32 + hv * 16));

#pragma unroll
    for (int nt = 0; nt < 4; ++nt)
#pragma unroll
      for (int r = 0; r < 8; ++r) o[nt][r] *= corr[r];

    const size_t vbase = (size_t)bh * HD * N_;
#pragma unroll
    for (int nt = 0; nt < 4; ++nt) {
      const v16bf vf = __builtin_bit_cast(
          v16bf,
          *(const v16s*)(Vt + vbase + (size_t)(nt * 16 + ln) * N_ + m0 +
                         hv * 16));
      o[nt] = __builtin_amdgcn_wmma_f32_16x16x32_bf16(
          false, pa, false, vf, (short)0, o[nt], false, false);
    }
  }

#pragma unroll
  for (int nt = 0; nt < 4; ++nt)
#pragma unroll
    for (int r = 0; r < 8; ++r) {
      const int row = q0 + hv * 8 + r;
      const int tok = bb * N_ + row;
      const int col = head * HD + nt * 16 + ln;
      O[(size_t)tok * DIM + col] = f2bf(o[nt][r] / lrow[r]);
    }
}

// ---------------------------------------------------------------------------
extern "C" void kernel_launch(void* const* d_in, const int* in_sizes, int n_in,
                              void* d_out, int out_size, void* d_ws,
                              size_t ws_size, hipStream_t stream) {
  const float* tgt   = (const float*)d_in[0];
  const float* src   = (const float*)d_in[1];
  const int* tpos    = (const int*)d_in[2];
  const int* spos    = (const int*)d_in[3];
  const float* caWq  = (const float*)d_in[4];
  const float* caWkv = (const float*)d_in[5];
  const float* caWo  = (const float*)d_in[6];
  const float* ca_qn = (const float*)d_in[7];
  const float* ca_kn = (const float*)d_in[8];
  const float* saWq  = (const float*)d_in[9];
  const float* saWkv = (const float*)d_in[10];
  const float* saWo  = (const float*)d_in[11];
  const float* sa_qn = (const float*)d_in[12];
  const float* sa_kn = (const float*)d_in[13];
  const float* pre_ca  = (const float*)d_in[14];
  const float* pre_sa  = (const float*)d_in[15];
  const float* pre_ffn = (const float*)d_in[16];
  const float* W1 = (const float*)d_in[17];
  const float* W2 = (const float*)d_in[18];
  const float* W3 = (const float*)d_in[19];

  char* p = (char*)d_ws;
  auto take = [&](size_t bytes) -> char* {
    char* r = p;
    p += (bytes + 255) & ~(size_t)255;
    return r;
  };
  float* x              = (float*)take((size_t)TOK * DIM * 4);
  unsigned short* wqbC  = (unsigned short*)take((size_t)DIM * DIM * 2);
  unsigned short* wkvbC = (unsigned short*)take((size_t)DIM * 2 * DIM * 2);
  unsigned short* wobC  = (unsigned short*)take((size_t)DIM * DIM * 2);
  unsigned short* wqbS  = (unsigned short*)take((size_t)DIM * DIM * 2);
  unsigned short* wkvbS = (unsigned short*)take((size_t)DIM * 2 * DIM * 2);
  unsigned short* wobS  = (unsigned short*)take((size_t)DIM * DIM * 2);
  unsigned short* w1b   = (unsigned short*)take((size_t)DIM * HID * 2);
  unsigned short* w2b   = (unsigned short*)take((size_t)HID * DIM * 2);
  unsigned short* w3b   = (unsigned short*)take((size_t)DIM * HID * 2);
  unsigned short* srcb  = (unsigned short*)take((size_t)TOK * DIM * 2);
  unsigned short* ybf   = (unsigned short*)take((size_t)TOK * DIM * 2);
  unsigned short* qbf   = (unsigned short*)take((size_t)TOK * DIM * 2);
  unsigned short* kbf   = (unsigned short*)take((size_t)TOK * DIM * 2);
  unsigned short* vbf   = (unsigned short*)take((size_t)TOK * DIM * 2);
  unsigned short* atnb  = (unsigned short*)take((size_t)TOK * DIM * 2);
  unsigned short* hb    = (unsigned short*)take((size_t)TOK * HID * 2);
  float* tmp1           = (float*)take((size_t)TOK * DIM * 4);
  float* tmp2           = (float*)take((size_t)TOK * HID * 4);

  (void)hipMemcpyAsync(x, tgt, (size_t)TOK * DIM * 4, hipMemcpyDeviceToDevice,
                       stream);

  // weights: convert + transpose to [N,K] bf16 for the NT GEMM / TDM path
  auto cvtT = [&](const float* s, unsigned short* d, int K, int N) {
    cvt_tr_k<<<dim3(1024), dim3(256), 0, stream>>>(s, d, K, N);
  };
  cvtT(caWq, wqbC, DIM, DIM);
  cvtT(caWkv, wkvbC, DIM, 2 * DIM);
  cvtT(caWo, wobC, DIM, DIM);
  cvtT(saWq, wqbS, DIM, DIM);
  cvtT(saWkv, wkvbS, DIM, 2 * DIM);
  cvtT(saWo, wobS, DIM, DIM);
  cvtT(W1, w1b, DIM, HID);
  cvtT(W2, w2b, HID, DIM);
  cvtT(W3, w3b, DIM, HID);
  cvt_bf16_k<<<dim3(1024), dim3(256), 0, stream>>>(src, srcb, TOK * DIM);

  auto gemm = [&](const unsigned short* A, const unsigned short* Bt, float* Cf,
                  unsigned short* Cb, int M, int N, int K, int mode) {
    gemm_bf16_k<<<dim3(N / 64, M / 128), dim3(256), 0, stream>>>(A, Bt, Cf, Cb,
                                                                 M, N, K, mode);
  };

  auto attn_block = [&](const unsigned short* yb, const unsigned short* kvin,
                        const unsigned short* wq, const unsigned short* wkv,
                        const unsigned short* wo, const float* qn,
                        const float* kn, const int* qp, const int* kp) {
    gemm(yb, wq, tmp1, nullptr, TOK, DIM, DIM, 0);
    gemm(kvin, wkv, tmp2, nullptr, TOK, 2 * DIM, DIM, 0);
    prep_qk_k<<<dim3(TOK), dim3(256), 0, stream>>>(tmp1, DIM, qn, qp, qbf, 0);
    prep_qk_k<<<dim3(TOK), dim3(256), 0, stream>>>(tmp2, 2 * DIM, kn, kp, kbf, 0);
    cvt_v_k<<<dim3((TOK * DIM + 255) / 256), dim3(256), 0, stream>>>(tmp2, vbf);
    flash_attn_k<<<dim3(N_ / 128, B_ * NH), dim3(256), 0, stream>>>(qbf, kbf,
                                                                    vbf, atnb);
    gemm(atnb, wo, x, nullptr, TOK, DIM, DIM, 2);  // residual accumulate
  };

  // cross-attention
  rmsnorm_rows_k<<<dim3(TOK), dim3(256), 0, stream>>>(x, pre_ca, ybf);
  attn_block(ybf, srcb, wqbC, wkvbC, wobC, ca_qn, ca_kn, tpos, spos);

  // self-attention
  rmsnorm_rows_k<<<dim3(TOK), dim3(256), 0, stream>>>(x, pre_sa, ybf);
  attn_block(ybf, ybf, wqbS, wkvbS, wobS, sa_qn, sa_kn, tpos, tpos);

  // FFN (SwiGLU)
  rmsnorm_rows_k<<<dim3(TOK), dim3(256), 0, stream>>>(x, pre_ffn, ybf);
  gemm(ybf, w1b, tmp2, nullptr, TOK, HID, DIM, 0);     // t1 = y@W1
  gemm(ybf, w3b, tmp2, hb, TOK, HID, DIM, 3);          // h = silu(t1)*(y@W3)
  gemm(hb, w2b, x, nullptr, TOK, DIM, HID, 2);         // x += h@W2

  (void)hipMemcpyAsync(d_out, x, (size_t)TOK * DIM * 4,
                       hipMemcpyDeviceToDevice, stream);
}
